// Decoder2_73993696575814
// MI455X (gfx1250) — compile-verified, matrix operand order checked by reading, not run on hardware
//
#include <hip/hip_runtime.h>
#include <hip/hip_fp16.h>

// Problem constants (match the reference; lengths is statically T)
#define B_  64
#define T_  1024
#define E_  256
#define H_  512
#define V_  1024
#define G3H (3 * H_)   // 1536

typedef __attribute__((ext_vector_type(16))) _Float16 v16h;
typedef __attribute__((ext_vector_type(8)))  _Float16 v8h;
typedef __attribute__((ext_vector_type(8)))  float    v8f;

#if __has_builtin(__builtin_amdgcn_tensor_load_to_lds)
#define HAVE_TDM 1
#else
#define HAVE_TDM 0
#endif

#if HAVE_TDM
typedef __attribute__((ext_vector_type(4))) unsigned int u32x4;
typedef __attribute__((ext_vector_type(8))) int          i32x8;
typedef __attribute__((ext_vector_type(4))) int          i32x4;

// 1-D contiguous TDM tile: nelem elements of (1<<dsz_code) bytes, global -> LDS.
// D# built per CDNA5 ISA 8.3/8.4: group0 {count=1, lds_addr, global_addr, type=2},
// group1 {data_size, tensor_dim0=tile_dim0=nelem, tensor_dim1=tile_dim1=1, stride0=nelem}.
// This toolchain's builtin takes 6 args (g0, g1, g2, g3, g4, cpol); unused
// groups are zero-filled (probe-verified form).
static __device__ __forceinline__ void tdm_load_1d(void* lds_ptr, const void* gptr,
                                                   unsigned nelem, unsigned dsz_code) {
    unsigned lds_off = (unsigned)(unsigned long long)(size_t)lds_ptr; // flat LDS addr low 32b = offset
    unsigned long long ga = (unsigned long long)(size_t)gptr;
    u32x4 g0;
    g0[0] = 1u;                                                  // count=1, user descriptor
    g0[1] = lds_off;                                             // lds_addr (bytes)
    g0[2] = (unsigned)ga;                                        // global_addr[31:0]
    g0[3] = (unsigned)((ga >> 32) & 0x01ffffffu) | (2u << 30);   // addr[56:32] | type=2
    i32x8 g1;
    g1[0] = (int)(dsz_code << 16);                               // data_size, mask=0
    g1[1] = (int)((nelem & 0xffffu) << 16);                      // tensor_dim0[15:0]
    g1[2] = (int)(((nelem >> 16) & 0xffffu) | (1u << 16));       // tensor_dim0[31:16], tensor_dim1=1
    g1[3] = (int)((nelem & 0xffffu) << 16);                      // tile_dim0 = nelem (<= 65535)
    g1[4] = 1;                                                   // tile_dim1 = 1
    g1[5] = (int)nelem;                                          // tensor_dim0_stride[31:0]
    g1[6] = 0;
    g1[7] = 0;
    i32x4 z4 = {0, 0, 0, 0};
    i32x8 z8 = {0, 0, 0, 0, 0, 0, 0, 0};
    __builtin_amdgcn_tensor_load_to_lds(g0, g1, z4, z4, z8, 0);
}

static __device__ __forceinline__ void tdm_wait0() {
#if __has_builtin(__builtin_amdgcn_s_wait_tensorcnt)
    __builtin_amdgcn_s_wait_tensorcnt(0);
#else
    asm volatile("s_wait_tensorcnt 0x0" ::: "memory");
#endif
}
#endif // HAVE_TDM

// ---------------------------------------------------------------------------
// WMMA 16x16x32 f16 tile loaders (wave32): lane row = lane&15, klo = 8*(lane>=16),
// elems 0..7 -> K=klo+0..7, elems 8..15 -> K=klo+16..23 (two contiguous 16B runs).
// Works for global or LDS source pointers (addrspace inferred).
// ---------------------------------------------------------------------------
static __device__ __forceinline__ v16h load_tile_f16(const _Float16* src,
                                                     int ld, int row0, int col0) {
    const int lane = threadIdx.x & 31;
    const int r    = lane & 15;
    const int klo  = (lane >> 4) << 3;
    const _Float16* p = src + (size_t)(row0 + r) * ld + col0 + klo;
    v8h lo = *(const v8h*)(p);
    v8h hi = *(const v8h*)(p + 16);
    return __builtin_shufflevector(lo, hi, 0,1,2,3,4,5,6,7,8,9,10,11,12,13,14,15);
}

static __device__ __forceinline__ v16h load_tile_f32_as_f16(const float* src,
                                                            int ld, int row0, int col0) {
    const int lane = threadIdx.x & 31;
    const int r    = lane & 15;
    const int klo  = (lane >> 4) << 3;
    const float* p = src + (size_t)(row0 + r) * ld + col0 + klo;
    v16h out;
#pragma unroll
    for (int i = 0; i < 8; ++i) {
        out[i]     = (_Float16)p[i];
        out[i + 8] = (_Float16)p[i + 16];
    }
    return out;
}

// ---------------------------------------------------------------------------
// One-time prep kernels
// ---------------------------------------------------------------------------
__global__ void cvt_f32_to_f16_kernel(const float* __restrict__ in,
                                      _Float16* __restrict__ out, int n) {
    int i = blockIdx.x * blockDim.x + threadIdx.x;
    if (i < n) out[i] = (_Float16)in[i];
}

__global__ void init_h_kernel(float* __restrict__ h32,
                              _Float16* __restrict__ h16a, _Float16* __restrict__ h16b) {
    int i = blockIdx.x * blockDim.x + threadIdx.x;   // B_*H_ threads
    h32[i]  = 0.0f;
    h16a[i] = (_Float16)0.0f;
    h16b[i] = (_Float16)0.0f;
}

// gx[b, j] = x[b,:] . W_ih[j,:] + b_ih[j]   (f32, one-time, tiny)
__global__ void gx_kernel(const float* __restrict__ x, const float* __restrict__ W_ih,
                          const float* __restrict__ b_ih, float* __restrict__ gx) {
    int i = blockIdx.x * blockDim.x + threadIdx.x;   // B_*G3H threads
    int b = i / G3H;
    int j = i - b * G3H;
    const float* xr = x + (size_t)b * E_;
    const float* wr = W_ih + (size_t)j * E_;
    float s = b_ih[j];
#pragma unroll 4
    for (int e = 0; e < E_; ++e) s = fmaf(xr[e], wr[e], s);
    gx[i] = s;
}

// ---------------------------------------------------------------------------
// Fused GRU step. One block per 16-column slice j0 = blockIdx.x*16 (32 blocks).
// 12 waves: wave w -> (gate g = w>>2, row tile tm = w&3); each computes the
// 16x16 pre-activation tile for its gate/columns with K=512 WMMA, A staged in
// LDS via TDM, B (W_hh) streamed from L2. Gate tiles land in LDS; then the
// block does the elementwise gate math + state update for its columns only
// (no cross-block hazard; h is ping-ponged in f16 across launches).
// ---------------------------------------------------------------------------
__global__ void __launch_bounds__(384) gru_step_kernel(const _Float16* __restrict__ h16_in,
                                                       const _Float16* __restrict__ Whh16,
                                                       const float* __restrict__ gx,
                                                       const float* __restrict__ b_hh,
                                                       float* __restrict__ h32,
                                                       _Float16* __restrict__ h16_out,
                                                       float* __restrict__ hidden_out,
                                                       int t) {
    __shared__ __align__(16) _Float16 hA[B_ * H_];      // 64 KB: full h state, f16
    __shared__ float gbuf[3][B_][16];                   // 12 KB: gate pre-activations

    const int tid = threadIdx.x;
    const int j0  = blockIdx.x * 16;

    // ---- stage h[64,512] f16 into LDS (async tensor DMA; wave 0 issues) ----
#if HAVE_TDM
    if (tid < 32) {
        tdm_load_1d(hA, h16_in, B_ * H_, /*f16*/1);
        tdm_wait0();
    }
#else
    for (int i = tid; i < (B_ * H_) / 8; i += blockDim.x)
        ((v8h*)hA)[i] = ((const v8h*)h16_in)[i];
#endif
    __syncthreads();

    // ---- WMMA: gate tile (g, tm) ----
    const int w  = tid >> 5;       // 0..11
    const int g  = w >> 2;         // gate 0..2
    const int tm = w & 3;          // row tile 0..3
    v8f acc = {};
    for (int k0 = 0; k0 < H_; k0 += 32) {
        v16h a = load_tile_f16(hA, H_, tm * 16, k0);                    // ds_load_b128
        v16h b = load_tile_f16(Whh16, H_, g * H_ + j0, k0);             // global (L2-hot)
        acc = __builtin_amdgcn_wmma_f32_16x16x32_f16(false, a, false, b,
                                                     (short)0, acc, false, false);
    }
    const int lane = tid & 31;
    const int n    = lane & 15;
    const int mh   = (lane >> 4) << 3;
#pragma unroll
    for (int r = 0; r < 8; ++r)
        gbuf[g][tm * 16 + mh + r][n] = acc[r];
    __syncthreads();

    // ---- gate math + state update for columns [j0, j0+16) ----
    for (int i = tid; i < B_ * 16; i += blockDim.x) {
        int b  = i >> 4;
        int jj = i & 15;
        int j  = j0 + jj;
        size_t rb = (size_t)b * G3H;
        float ghr = gbuf[0][b][jj];
        float ghz = gbuf[1][b][jj];
        float ghn = gbuf[2][b][jj];
        float rg = 1.0f / (1.0f + __expf(-(gx[rb + j]          + ghr + b_hh[j])));
        float zg = 1.0f / (1.0f + __expf(-(gx[rb + H_ + j]     + ghz + b_hh[H_ + j])));
        float ng = tanhf(gx[rb + 2 * H_ + j] + rg * (ghn + b_hh[2 * H_ + j]));
        int hidx = b * H_ + j;
        float hnew = (1.0f - zg) * ng + zg * h32[hidx];
        h32[hidx]     = hnew;
        h16_out[hidx] = (_Float16)hnew;
        hidden_out[((size_t)b * T_ + t) * H_ + j] = hnew;
    }
}

// ---------------------------------------------------------------------------
// Final: logp[bt, v] = log_softmax_v( hidden[bt,:] @ W_fc[v,:]^T )
// One block per 16-row slab (4096 blocks, 8 waves). Hidden slab (16x512 f32,
// contiguous 32 KB) staged into LDS via TDM; W_fc f16 (1 MB) streams from L2.
// Logits stay in registers (8 x v8f per wave); softmax via 16-lane shfl_xor
// reductions + tiny LDS cross-wave reduce.
// ---------------------------------------------------------------------------
__global__ void __launch_bounds__(256) fc_logsoftmax_kernel(const float* __restrict__ hidden,
                                                            const _Float16* __restrict__ Wfc16,
                                                            float* __restrict__ logp) {
    __shared__ __align__(16) float sA[16 * H_];   // 32 KB hidden slab (f32)
    __shared__ float redbuf[8][16];
    __shared__ float rowred[16];

    const int tid  = threadIdx.x;
    const int lane = tid & 31;
    const int wave = tid >> 5;
    const int hi   = lane >> 4;            // 0: rows 0-7, 1: rows 8-15
    const int bt0  = blockIdx.x * 16;

#if HAVE_TDM
    if (tid < 32) {
        tdm_load_1d(sA, hidden + (size_t)bt0 * H_, 16 * H_, /*f32*/2);
        tdm_wait0();
    }
#else
    {
        const float* src = hidden + (size_t)bt0 * H_;
        for (int i = tid; i < (16 * H_) / 4; i += blockDim.x)
            ((float4*)sA)[i] = ((const float4*)src)[i];
    }
#endif
    __syncthreads();

    v8f acc[8];
#pragma unroll
    for (int nt = 0; nt < 8; ++nt) acc[nt] = (v8f){};

    for (int k0 = 0; k0 < H_; k0 += 32) {
        v16h a = load_tile_f32_as_f16(sA, H_, 0, k0);                   // LDS, cvt to f16
#pragma unroll
        for (int nt = 0; nt < 8; ++nt) {
            v16h b = load_tile_f16(Wfc16, H_, (wave * 8 + nt) * 16, k0);
            acc[nt] = __builtin_amdgcn_wmma_f32_16x16x32_f16(false, a, false, b,
                                                             (short)0, acc[nt], false, false);
        }
    }

    // ---- row max ----
    float pm[8];
#pragma unroll
    for (int r = 0; r < 8; ++r) {
        float m = acc[0][r];
#pragma unroll
        for (int nt = 1; nt < 8; ++nt) m = fmaxf(m, acc[nt][r]);
#pragma unroll
        for (int s = 1; s < 16; s <<= 1) m = fmaxf(m, __shfl_xor(m, s, 32));
        pm[r] = m;
    }
    if ((lane & 15) == 0) {
#pragma unroll
        for (int r = 0; r < 8; ++r) redbuf[wave][hi * 8 + r] = pm[r];
    }
    __syncthreads();
    if (tid < 16) {
        float m = redbuf[0][tid];
#pragma unroll
        for (int ww = 1; ww < 8; ++ww) m = fmaxf(m, redbuf[ww][tid]);
        rowred[tid] = m;
    }
    __syncthreads();
    float rmax[8];
#pragma unroll
    for (int r = 0; r < 8; ++r) rmax[r] = rowred[hi * 8 + r];
    __syncthreads();

    // ---- row sum(exp) ----
    float ps[8];
#pragma unroll
    for (int r = 0; r < 8; ++r) {
        float s = 0.0f;
#pragma unroll
        for (int nt = 0; nt < 8; ++nt) s += __expf(acc[nt][r] - rmax[r]);
#pragma unroll
        for (int sh = 1; sh < 16; sh <<= 1) s += __shfl_xor(s, sh, 32);
        ps[r] = s;
    }
    if ((lane & 15) == 0) {
#pragma unroll
        for (int r = 0; r < 8; ++r) redbuf[wave][hi * 8 + r] = ps[r];
    }
    __syncthreads();
    if (tid < 16) {
        float s = redbuf[0][tid];
#pragma unroll
        for (int ww = 1; ww < 8; ++ww) s += redbuf[ww][tid];
        rowred[tid] = s;
    }
    __syncthreads();

    float lse[8];
#pragma unroll
    for (int r = 0; r < 8; ++r) lse[r] = rmax[r] + __logf(rowred[hi * 8 + r]);

    // ---- write log-probs ----
    const int n = lane & 15;
#pragma unroll
    for (int nt = 0; nt < 8; ++nt) {
        int col = wave * 128 + nt * 16 + n;
#pragma unroll
        for (int r = 0; r < 8; ++r) {
            int row = hi * 8 + r;
            logp[(size_t)(bt0 + row) * V_ + col] = acc[nt][r] - lse[r];
        }
    }
}

// ---------------------------------------------------------------------------
// Host launcher
// Workspace layout (bytes):
//   [0x000000) Whh16 : 1536*512 f16 = 1,572,864
//   [0x180000) Wfc16 : 1024*512 f16 = 1,048,576
//   [0x280000) gx    : 64*1536  f32 =   393,216
//   [0x2E0000) h32   : 64*512   f32 =   131,072
//   [0x300000) h16A  : 64*512   f16 =    65,536
//   [0x310000) h16B  : 64*512   f16 =    65,536
// total ~3.13 MB
// ---------------------------------------------------------------------------
extern "C" void kernel_launch(void* const* d_in, const int* in_sizes, int n_in,
                              void* d_out, int out_size, void* d_ws, size_t ws_size,
                              hipStream_t stream) {
    (void)in_sizes; (void)n_in; (void)out_size; (void)ws_size;

    const float* x     = (const float*)d_in[0];  // encoder_embedding [B,E]
    // d_in[1] = y (int, unused by the math); d_in[2] = lengths (== T_, static)
    const float* W_ih  = (const float*)d_in[3];  // [3H,E]
    const float* W_hh  = (const float*)d_in[4];  // [3H,H]
    const float* b_ih  = (const float*)d_in[5];  // [3H]
    const float* b_hh  = (const float*)d_in[6];  // [3H]
    const float* W_fc  = (const float*)d_in[7];  // [V,H]

    float* out        = (float*)d_out;
    float* hidden_out = out;                                   // [B,T,H]
    float* logp       = out + (size_t)B_ * T_ * H_;            // [B,T,V]

    char* ws = (char*)d_ws;
    _Float16* Whh16 = (_Float16*)(ws + 0x000000);
    _Float16* Wfc16 = (_Float16*)(ws + 0x180000);
    float*    gx    = (float*)   (ws + 0x280000);
    float*    h32   = (float*)   (ws + 0x2E0000);
    _Float16* h16A  = (_Float16*)(ws + 0x300000);
    _Float16* h16B  = (_Float16*)(ws + 0x310000);

    // one-time prep
    cvt_f32_to_f16_kernel<<<(G3H * H_ + 255) / 256, 256, 0, stream>>>(W_hh, Whh16, G3H * H_);
    cvt_f32_to_f16_kernel<<<(V_ * H_ + 255) / 256, 256, 0, stream>>>(W_fc, Wfc16, V_ * H_);
    init_h_kernel<<<(B_ * H_) / 256, 256, 0, stream>>>(h32, h16A, h16B);
    gx_kernel<<<(B_ * G3H) / 256, 256, 0, stream>>>(x, W_ih, b_ih, gx);

    // sequential GRU recurrence: one fused kernel per step, h16 ping-pong
    for (int t = 0; t < T_; ++t) {
        const _Float16* hin  = (t & 1) ? h16B : h16A;
        _Float16*       hout = (t & 1) ? h16A : h16B;
        gru_step_kernel<<<G3H / 48, 384, 0, stream>>>(hin, Whh16, gx, b_hh,
                                                      h32, hout, hidden_out, t);
    }

    // big GEMM + fused log_softmax
    fc_logsoftmax_kernel<<<(B_ * T_) / 16, 256, 0, stream>>>(hidden_out, Wfc16, logp);
}
